// STSN_Offset_36601711296534
// MI455X (gfx1250) — matrix operand order repaired and checked by reading, not compiled, not run on hardware
//
#include <hip/hip_runtime.h>

// ---------------------------------------------------------------------------
// Problem constants (reference: B=4, C=256 feat channels, H=W=64, 3x3 taps)
// feat is kept NHWC so implicit-GEMM K (channels) is contiguous in memory.
// K ordering is TAP-MAJOR: kp = tap*256 + c  (each 32-chunk has one tap).
// ---------------------------------------------------------------------------
#define Bn   4
#define Cn   256
#define Hn   64
#define Wn   64
#define HWn  (Hn * Wn)          // 4096
#define KTAP 9
#define KKn  (Cn * KTAP)        // 2304 = implicit-GEMM K
#define NCHUNK (KKn / 32)       // 72 chunks of K=32 for bf16 WMMA

typedef __attribute__((ext_vector_type(16))) __bf16 v16bf;
typedef __attribute__((ext_vector_type(8)))  __bf16 v8bf;
typedef __attribute__((ext_vector_type(4)))  __bf16 v4bf;
typedef __attribute__((ext_vector_type(8)))  float  v8f;
typedef __attribute__((ext_vector_type(4)))  float  v4f;

// ---------------------------------------------------------------------------
// concat(x,y) NCHW -> feat NHWC, LDS-tiled transpose (coalesced both sides)
// Grid = B * (HW/32) * (C/32) = 4*128*8 = 4096 blocks of 256.
// ---------------------------------------------------------------------------
__global__ __launch_bounds__(256) void concat_nhwc(
    const float* __restrict__ x, const float* __restrict__ y,
    float* __restrict__ feat) {
  __shared__ float tile[32][33];
  const int bid = blockIdx.x;
  const int cb  = (bid & 7) << 5;           // channel block
  const int pb  = ((bid >> 3) & 127) << 5;  // pixel block
  const int b   = bid >> 10;
  const int tr  = threadIdx.x >> 5;         // 0..7
  const int tc  = threadIdx.x & 31;
  #pragma unroll
  for (int j = 0; j < 4; ++j) {
    int c = cb + tr + j * 8;
    const float* src = (c < 128)
        ? (x + (((size_t)b * 128 + c) << 12))
        : (y + (((size_t)b * 128 + (c - 128)) << 12));
    tile[tr + j * 8][tc] = src[pb + tc];
  }
  __syncthreads();
  #pragma unroll
  for (int j = 0; j < 4; ++j) {
    int prow = tr + j * 8;
    feat[((size_t)b * HWn + pb + prow) * Cn + cb + tc] = tile[tc][prow];
  }
}

// ---------------------------------------------------------------------------
// deform_w f32 [3][N=256][C=256][9] -> bf16 [3][N][kp = tap*256+c]
// ---------------------------------------------------------------------------
__global__ __launch_bounds__(256) void cvt_deform_w(
    const float* __restrict__ src, __bf16* __restrict__ dst) {
  int i  = blockIdx.x * 256 + threadIdx.x;   // 3*256*2304 = 1,769,472
  int kp = i % KKn;
  int n  = (i / KKn) & 255;
  int l  = i / (KKn * 256);
  int tap = kp >> 8;
  int c   = kp & 255;
  dst[i] = (__bf16)src[(((size_t)l * 256 + n) * 256 + c) * 9 + tap];
}

// offset_w f32 [4][18][256][9] -> bf16 [4][32][kp], rows 18..31 zero-padded
__global__ __launch_bounds__(256) void cvt_offset_w(
    const float* __restrict__ src, __bf16* __restrict__ dst) {
  int i  = blockIdx.x * 256 + threadIdx.x;   // 4*32*2304 = 294,912
  int kp = i % KKn;
  int n  = (i / KKn) & 31;
  int l  = i / (KKn * 32);
  int tap = kp >> 8;
  int c   = kp & 255;
  float v = (n < 18) ? src[(((size_t)l * 18 + n) * 256 + c) * 9 + tap] : 0.0f;
  dst[i] = (__bf16)v;
}

// ---------------------------------------------------------------------------
// Plain 3x3 conv (stride1 pad1), feat NHWC -> out NCHW (outC<=32), WMMA.
// Block = 64 threads (2 waves), each wave one 16-wide N tile. Grid = 1024.
// ---------------------------------------------------------------------------
__global__ __launch_bounds__(64) void conv3x3_wmma(
    const float* __restrict__ feat, const __bf16* __restrict__ wB,
    float* __restrict__ out, int outC) {
  __shared__ __align__(16) __bf16 Abuf[2][16][48];   // 32 used + pad

  const int bid = blockIdx.x;
  const int w0  = (bid & 3) << 4;
  const int h   = (bid >> 2) & (Hn - 1);
  const int b   = bid >> 8;

  const int tid  = threadIdx.x;
  const int lane = tid & 31;
  const int wv   = tid >> 5;          // 0..1 -> N tile
  const int row  = lane & 15;
  const int hi   = lane >> 4;

  // A-builder role: pixel am, 8 consecutive channels starting at kkb
  const int am  = tid >> 2;           // 0..15
  const int kkb = (tid & 3) << 3;     // 0,8,16,24

  v8f acc = {};
  const __bf16* bbase = wB + (size_t)(wv * 16 + row) * KKn + hi * 16;
  const float*  fbase = feat + (size_t)b * HWn * Cn;

  for (int tap = 0; tap < KTAP; ++tap) {
    const int ty = tap / 3 - 1;
    const int tx = tap - (tap / 3) * 3 - 1;
    const int yy = h + ty;
    const int xx = w0 + am + tx;
    const bool inb = ((unsigned)yy < (unsigned)Hn) &&
                     ((unsigned)xx < (unsigned)Wn);
    const float* pl = fbase + ((size_t)yy * Wn + xx) * Cn + kkb;

    #pragma unroll
    for (int cc = 0; cc < 8; ++cc) {
      const int kc = tap * 8 + cc;
      const int p  = kc & 1;
      v8bf vals;
      if (inb) {
        v4f f0 = *(const v4f*)(pl + (cc << 5));
        v4f f1 = *(const v4f*)(pl + (cc << 5) + 4);
        #pragma unroll
        for (int j = 0; j < 4; ++j) {
          vals[j]     = (__bf16)f0[j];
          vals[j + 4] = (__bf16)f1[j];
        }
      } else {
        #pragma unroll
        for (int j = 0; j < 8; ++j) vals[j] = (__bf16)0.0f;
      }
      *(v8bf*)&Abuf[p][am][kkb] = vals;
      __syncthreads();

      v8bf alo = *(const v8bf*)&Abuf[p][row][hi * 16];
      v8bf ahi = *(const v8bf*)&Abuf[p][row][hi * 16 + 8];
      v16bf afrag = __builtin_shufflevector(alo, ahi,
          0, 1, 2, 3, 4, 5, 6, 7, 8, 9, 10, 11, 12, 13, 14, 15);
      v16bf bfrag = *(const v16bf*)(bbase + (size_t)kc * 32);
      __builtin_prefetch(bbase + (size_t)(kc + 1) * 32, 0, 0);

      acc = __builtin_amdgcn_wmma_f32_16x16x32_bf16(
          false, afrag, false, bfrag, (short)0, acc, false, false);
    }
  }

  // ---- store C (NCHW): VGPR r -> M=r / r+8; N = lane&15 ----
  int nOut = wv * 16 + row;
  if (nOut < outC) {
    int mbase = hi * 8;
    for (int r = 0; r < 8; ++r)
      out[(((size_t)b * outC + nOut) * Hn + h) * Wn + w0 + mbase + r] = acc[r];
  }
}

// ---------------------------------------------------------------------------
// Deformable 3x3 conv 256 -> 256, feat NHWC -> out NHWC, WMMA implicit GEMM.
// Block = 256 threads (8 waves). M-tile = 32 px, 2 N-tiles/wave -> 4 WMMA.
// Grid = B*H*(W/32) = 512.
// ---------------------------------------------------------------------------
struct TapInfo {
  int   i00, i01, i10, i11;
  float w00, w01, w10, w11;
};

__global__ __launch_bounds__(256) void deform_wmma(
    const float* __restrict__ feat, const float* __restrict__ off,
    const __bf16* __restrict__ wB, float* __restrict__ out) {
  __shared__ __align__(16) __bf16 Abuf[2][32][48];
  __shared__ TapInfo taps[32 * KTAP];

  const int bid = blockIdx.x;
  const int w0  = (bid & 1) << 5;
  const int h   = (bid >> 1) & (Hn - 1);
  const int b   = bid >> 7;

  const int tid  = threadIdx.x;
  const int lane = tid & 31;
  const int wv   = tid >> 5;          // 0..7
  const int row  = lane & 15;
  const int hi   = lane >> 4;

  // ---- per-pixel / per-tap bilinear precompute (32 px x 9 taps) ----
  for (int t = tid; t < 32 * KTAP; t += 256) {
    int m   = t / KTAP;
    int tap = t - m * KTAP;
    int ty  = tap / 3 - 1;
    int tx  = tap - (tap / 3) * 3 - 1;
    int xpix = w0 + m;
    size_t obase = (((size_t)b * 18 + tap * 2) * Hn + h) * Wn + xpix;  // NCHW
    float dy = off[obase];
    float dx = off[obase + (size_t)HWn];
    float py = dy + (float)ty + (float)h;
    float px = dx + (float)tx + (float)xpix;
    float y0f = floorf(py), x0f = floorf(px);
    float wy1 = py - y0f, wy0 = 1.0f - wy1;
    float wx1 = px - x0f, wx0 = 1.0f - wx1;
    int y0 = (int)y0f, x0 = (int)x0f;
    int y1 = y0 + 1,   x1 = x0 + 1;
    float vy0 = (y0 >= 0 && y0 < Hn) ? 1.0f : 0.0f;
    float vy1 = (y1 >= 0 && y1 < Hn) ? 1.0f : 0.0f;
    float vx0 = (x0 >= 0 && x0 < Wn) ? 1.0f : 0.0f;
    float vx1 = (x1 >= 0 && x1 < Wn) ? 1.0f : 0.0f;
    int y0c = y0 < 0 ? 0 : (y0 > Hn - 1 ? Hn - 1 : y0);
    int y1c = y1 < 0 ? 0 : (y1 > Hn - 1 ? Hn - 1 : y1);
    int x0c = x0 < 0 ? 0 : (x0 > Wn - 1 ? Wn - 1 : x0);
    int x1c = x1 < 0 ? 0 : (x1 > Wn - 1 ? Wn - 1 : x1);
    TapInfo ti;
    ti.i00 = y0c * Wn + x0c;  ti.i01 = y0c * Wn + x1c;
    ti.i10 = y1c * Wn + x0c;  ti.i11 = y1c * Wn + x1c;
    ti.w00 = wy0 * wx0 * vy0 * vx0;
    ti.w01 = wy0 * wx1 * vy0 * vx1;
    ti.w10 = wy1 * wx0 * vy1 * vx0;
    ti.w11 = wy1 * wx1 * vy1 * vx1;
    taps[t] = ti;
  }
  __syncthreads();

  // A-builder role: pixel am, 4 consecutive channels starting at kkb
  const int am  = tid >> 3;            // 0..31
  const int kkb = (tid & 7) << 2;      // 0,4,...,28
  const TapInfo* tp = &taps[am * KTAP];

  v8f acc00 = {}, acc01 = {}, acc10 = {}, acc11 = {};
  const __bf16* b0base = wB + (size_t)(wv * 32 + row) * KKn + hi * 16;
  const __bf16* b1base = b0base + (size_t)16 * KKn;
  const float*  fbase  = feat + (size_t)b * HWn * Cn;

  for (int tap = 0; tap < KTAP; ++tap) {
    TapInfo ti = tp[tap];
    const float* p00 = fbase + (size_t)ti.i00 * Cn + kkb;
    const float* p01 = fbase + (size_t)ti.i01 * Cn + kkb;
    const float* p10 = fbase + (size_t)ti.i10 * Cn + kkb;
    const float* p11 = fbase + (size_t)ti.i11 * Cn + kkb;

    #pragma unroll
    for (int cc = 0; cc < 8; ++cc) {
      const int kc = tap * 8 + cc;
      const int p  = kc & 1;
      const int cb = cc << 5;
      v4f f00 = *(const v4f*)(p00 + cb);
      v4f f01 = *(const v4f*)(p01 + cb);
      v4f f10 = *(const v4f*)(p10 + cb);
      v4f f11 = *(const v4f*)(p11 + cb);
      v4bf vals;
      #pragma unroll
      for (int j = 0; j < 4; ++j)
        vals[j] = (__bf16)(ti.w00 * f00[j] + ti.w01 * f01[j] +
                           ti.w10 * f10[j] + ti.w11 * f11[j]);
      *(v4bf*)&Abuf[p][am][kkb] = vals;
      __syncthreads();

      v8bf a0lo = *(const v8bf*)&Abuf[p][row][hi * 16];
      v8bf a0hi = *(const v8bf*)&Abuf[p][row][hi * 16 + 8];
      v16bf a0 = __builtin_shufflevector(a0lo, a0hi,
          0, 1, 2, 3, 4, 5, 6, 7, 8, 9, 10, 11, 12, 13, 14, 15);
      v8bf a1lo = *(const v8bf*)&Abuf[p][row + 16][hi * 16];
      v8bf a1hi = *(const v8bf*)&Abuf[p][row + 16][hi * 16 + 8];
      v16bf a1 = __builtin_shufflevector(a1lo, a1hi,
          0, 1, 2, 3, 4, 5, 6, 7, 8, 9, 10, 11, 12, 13, 14, 15);
      v16bf bf0 = *(const v16bf*)(b0base + (size_t)kc * 32);
      v16bf bf1 = *(const v16bf*)(b1base + (size_t)kc * 32);
      __builtin_prefetch(b0base + (size_t)(kc + 1) * 32, 0, 0);
      __builtin_prefetch(b1base + (size_t)(kc + 1) * 32, 0, 0);

      acc00 = __builtin_amdgcn_wmma_f32_16x16x32_bf16(
          false, a0, false, bf0, (short)0, acc00, false, false);
      acc01 = __builtin_amdgcn_wmma_f32_16x16x32_bf16(
          false, a0, false, bf1, (short)0, acc01, false, false);
      acc10 = __builtin_amdgcn_wmma_f32_16x16x32_bf16(
          false, a1, false, bf0, (short)0, acc10, false, false);
      acc11 = __builtin_amdgcn_wmma_f32_16x16x32_bf16(
          false, a1, false, bf1, (short)0, acc11, false, false);
    }
  }

  // ---- store four 16x16 C tiles, NHWC (lane-contiguous in n) ----
  int mbase = hi * 8;
  int n0 = wv * 32 + row;
  int n1 = n0 + 16;
  size_t pixbase = ((size_t)b * Hn + h) * Wn + w0 + mbase;
  for (int r = 0; r < 8; ++r) {
    size_t q0 = (pixbase + r) * Cn;
    size_t q1 = (pixbase + 16 + r) * Cn;
    out[q0 + n0] = acc00[r];
    out[q0 + n1] = acc01[r];
    out[q1 + n0] = acc10[r];
    out[q1 + n1] = acc11[r];
  }
}

// ---------------------------------------------------------------------------
// Host-side orchestration
// ---------------------------------------------------------------------------
extern "C" void kernel_launch(void* const* d_in, const int* in_sizes, int n_in,
                              void* d_out, int out_size, void* d_ws, size_t ws_size,
                              hipStream_t stream) {
  const float* x        = (const float*)d_in[0];
  const float* y        = (const float*)d_in[1];
  const float* offset_w = (const float*)d_in[2];   // [4][18][256][3][3]
  const float* deform_w = (const float*)d_in[3];   // [3][256][256][3][3]
  float* out = (float*)d_out;                      // [4][18][64][64] NCHW

  char* ws = (char*)d_ws;
  const size_t F = (size_t)Bn * Cn * HWn * sizeof(float);       // 16 MB
  float*  featA  = (float*)(ws);                                 // NHWC
  float*  featB  = (float*)(ws + F);                             // NHWC
  float*  offBuf = (float*)(ws + 2 * F);                         // NCHW, 1.2 MB
  __bf16* offWb  = (__bf16*)(ws + 2 * F + (2u << 20));           // [4][32][2304]
  __bf16* dWb    = (__bf16*)(ws + 2 * F + (3u << 20));           // [3][256][2304]

  concat_nhwc<<<Bn * (HWn / 32) * (Cn / 32), 256, 0, stream>>>(x, y, featA);
  cvt_offset_w<<<(4 * 32 * KKn) / 256, 256, 0, stream>>>(offset_w, offWb);
  cvt_deform_w<<<(3 * Cn * KKn) / 256, 256, 0, stream>>>(deform_w, dWb);

  const int cblocks = Bn * Hn * (Wn / 16);   // 1024
  const int dblocks = Bn * Hn * (Wn / 32);   // 512
  float* cur = featA;
  float* nxt = featB;
  for (int i = 0; i < 3; ++i) {
    conv3x3_wmma<<<cblocks, 64, 0, stream>>>(
        cur, offWb + (size_t)i * 32 * KKn, offBuf, 18);
    deform_wmma<<<dblocks, 256, 0, stream>>>(
        cur, offBuf, dWb + (size_t)i * Cn * KKn, nxt);
    float* t = cur; cur = nxt; nxt = t;
  }
  conv3x3_wmma<<<cblocks, 64, 0, stream>>>(
      cur, offWb + (size_t)3 * 32 * KKn, out, 18);
}
